// SelfAttentionBlock_7481833029936
// MI455X (gfx1250) — compile-verified, matrix operand order checked by reading, not run on hardware
//
#include <hip/hip_runtime.h>
#include <hip/hip_bf16.h>
#include <math.h>

// ---------------------------------------------------------------------------
// Problem constants (from reference): B=2, C=768, T=8, H=W=24, nh=12, e=64
// ---------------------------------------------------------------------------
#define CB    2
#define CC    768
#define CT    8
#define CH    24
#define CW    24
#define CNH   12
#define CE    64
#define CS    576            // H*W
#define CN    16             // B*T
#define CROWS 9216           // CN*CS
#define CQKV  2304           // 3*C

typedef __attribute__((ext_vector_type(16))) __bf16 v16bf;
typedef __attribute__((ext_vector_type(8)))  __bf16 v8bf;
typedef __attribute__((ext_vector_type(4)))  __bf16 v4bf;
typedef __attribute__((ext_vector_type(8)))  float  v8f;
typedef __attribute__((ext_vector_type(4)))  unsigned int u32x4;
typedef __attribute__((ext_vector_type(8)))  int i32x8;
typedef __attribute__((ext_vector_type(4)))  int i32x4;

// CDNA5 feature probes (device pass only; host pass always takes fallbacks)
#if defined(__HIP_DEVICE_COMPILE__)
# if __has_builtin(__builtin_amdgcn_global_load_async_to_lds_b128)
#  define HAVE_ASYNC_LDS 1
# endif
# if __has_builtin(__builtin_amdgcn_tensor_load_to_lds)
#  define HAVE_TDM 1
# endif
#endif

__device__ __forceinline__ v16bf cat8(v8bf lo, v8bf hi) {
  return __builtin_shufflevector(lo, hi, 0,1,2,3,4,5,6,7,8,9,10,11,12,13,14,15);
}

__device__ __forceinline__ unsigned short f2h_bits(float f) {
  _Float16 h = (_Float16)f;
  return __builtin_bit_cast(unsigned short, h);
}
__device__ __forceinline__ float h_bits2f(unsigned short u) {
  return (float)__builtin_bit_cast(_Float16, u);
}
__device__ __forceinline__ unsigned short f2bf_bits(float f) {
  __bf16 h = (__bf16)f;
  return __builtin_bit_cast(unsigned short, h);
}

// 16B global -> LDS copy: async on CDNA5 (ASYNCcnt), register bounce otherwise.
// Builtin signature (from hipcc diagnostic): (i32x4 AS1*, i32x4 AS3*, imm, imm)
__device__ __forceinline__ void g2l_b128(__bf16* l, const __bf16* g) {
#ifdef HAVE_ASYNC_LDS
  __builtin_amdgcn_global_load_async_to_lds_b128(
      (__attribute__((address_space(1))) i32x4*)g,
      (__attribute__((address_space(3))) i32x4*)l, 0, 0);
#else
  *(v8bf*)l = *(const v8bf*)g;
#endif
}

#ifdef HAVE_ASYNC_LDS
#define WAIT_ASYNC_2() asm volatile("s_wait_asynccnt 0x2" ::: "memory")
#define WAIT_ASYNC_0() asm volatile("s_wait_asynccnt 0x0" ::: "memory")
#else
#define WAIT_ASYNC_2() do {} while (0)
#define WAIT_ASYNC_0() do {} while (0)
#endif

#ifdef HAVE_TDM
// 2-D TDM tile load: global (row-major, stride0 elems/row) -> LDS.
// d0_flags carries data_size + pad config bits of D# group1 dword0.
__device__ __forceinline__ void tdm_load_2d(unsigned lds_off, const void* gptr,
    unsigned tensor_d0, unsigned tensor_d1, unsigned tile_d0, unsigned tile_d1,
    unsigned stride0, unsigned d0_flags)
{
  unsigned long long ga = (unsigned long long)(uintptr_t)gptr;
  u32x4 g0;
  g0[0] = 1u;                                            // count=1 (valid)
  g0[1] = lds_off;                                       // lds_addr
  g0[2] = (unsigned)(ga & 0xffffffffull);                // global_addr lo
  g0[3] = (unsigned)((ga >> 32) & 0x1ffffffull) | 0x80000000u;  // hi | type=2
  i32x8 g1;
  g1[0] = (int)d0_flags;                                 // mask=0|data_size|pad
  g1[1] = (int)(tensor_d0 << 16);                        // tensor_dim0 lo16
  g1[2] = (int)((tensor_d0 >> 16) | (tensor_d1 << 16));  // d0 hi | d1 lo
  g1[3] = (int)((tensor_d1 >> 16) | (tile_d0 << 16));    // d1 hi | tile_dim0
  g1[4] = (int)(tile_d1 & 0xffffu);                      // tile_dim1 | dim2=0
  g1[5] = (int)stride0;                                  // dim0_stride lo32
  g1[6] = 0;                                             // stride hi | dim1_stride
  g1[7] = 0;
  i32x4 z4; z4[0] = z4[1] = z4[2] = z4[3] = 0;
# if __clang_major__ >= 23
  i32x8 z8;
  #pragma unroll
  for (int i = 0; i < 8; ++i) z8[i] = 0;
  __builtin_amdgcn_tensor_load_to_lds(g0, g1, z4, z4, z8, 0);
# else
  __builtin_amdgcn_tensor_load_to_lds(g0, g1, z4, z4, 0);
# endif
}
#endif

// ---------------------------------------------------------------------------
// K0: one-time weight conversion f32 -> bf16 (removes cvt chains from GEMMs)
// ---------------------------------------------------------------------------
__global__ __launch_bounds__(256)
void k_cvt_w(const float* __restrict__ qw, const float* __restrict__ ow,
             __bf16* __restrict__ qwb, __bf16* __restrict__ owb)
{
  size_t i = (size_t)blockIdx.x * 256 + threadIdx.x;
  if (i < (size_t)CQKV * CC) qwb[i] = (__bf16)qw[i];
  if (i < (size_t)CC * CC)   owb[i] = (__bf16)ow[i];
}

// ---------------------------------------------------------------------------
// K1: cs[b,c] = cond[b,:] . norm_w[c,:] + 1     (tiny GEMV, 1536 outputs)
// ---------------------------------------------------------------------------
__global__ __launch_bounds__(256)
void k_cond_scale(const float* __restrict__ cond, const float* __restrict__ nw,
                  float* __restrict__ cs)
{
  int gid = blockIdx.x * 256 + threadIdx.x;
  if (gid >= CB * CC) return;
  int b = gid / CC, c = gid % CC;
  const float* cr = cond + (size_t)b * CC;
  const float* wr = nw   + (size_t)c * CC;
  float s = 0.f;
  #pragma unroll 4
  for (int k = 0; k < CC; ++k) s = fmaf(cr[k], wr[k], s);
  cs[gid] = s + 1.f;
}

// ---------------------------------------------------------------------------
// K2: RMS-norm per spatial row -> xn (bf16, row-major [row, C])
// ---------------------------------------------------------------------------
__global__ __launch_bounds__(256)
void k_rmsnorm(const float* __restrict__ x, const float* __restrict__ cs,
               __bf16* __restrict__ xn)
{
  __shared__ float red[256];
  const int r   = blockIdx.x;
  const int tid = threadIdx.x;
  const int b   = r / (CT * CS);
  const int t   = (r / CS) % CT;
  const int si  = r % CS;
  float vals[3];
  float ssum = 0.f;
  #pragma unroll
  for (int i = 0; i < 3; ++i) {
    int c = tid + i * 256;
    float v = x[(((size_t)b * CC + c) * CT + t) * CS + si];
    vals[i] = v;
    ssum += v * v;
  }
  red[tid] = ssum;
  __syncthreads();
  for (int st = 128; st > 0; st >>= 1) {
    if (tid < st) red[tid] += red[tid + st];
    __syncthreads();
  }
  float inv = rsqrtf(red[0] * (1.0f / 768.0f) + 1e-6f);
  #pragma unroll
  for (int i = 0; i < 3; ++i) {
    int c = tid + i * 256;
    xn[(size_t)r * CC + c] = (__bf16)(vals[i] * cs[b * CC + c] * inv);
  }
}

// ---------------------------------------------------------------------------
// Shared GEMM machinery: 8 waves, 64x64 C tile, K-step 32, double-buffered
// LDS staging of A(64x32 bf16) and B(64x32 bf16) via async-to-LDS.
// ---------------------------------------------------------------------------
__device__ __forceinline__ void stage_tiles(const __bf16* __restrict__ A,
                                            const __bf16* __restrict__ Bw,
                                            int row0b, int col0b, int kk,
                                            __bf16 (*Al)[32], __bf16 (*Bl)[32],
                                            int tid)
{
  const int row = tid >> 2;
  const int c8  = (tid & 3) * 8;
  g2l_b128(&Al[row][c8], A  + (size_t)(row0b + row) * CC + kk + c8);
  g2l_b128(&Bl[row][c8], Bw + (size_t)(col0b + row) * CC + kk + c8);
}

// ---------------------------------------------------------------------------
// K3: QKV GEMM (NT): qkv[r,j] = xn[r,:] . qkv_w[j,:]; scatter to q/k/v bf16
// ---------------------------------------------------------------------------
__global__ __launch_bounds__(256)
void k_qkv_gemm(const __bf16* __restrict__ xn, const __bf16* __restrict__ wq,
                __bf16* __restrict__ qd, __bf16* __restrict__ kd,
                __bf16* __restrict__ vd)
{
  __shared__ __align__(16) __bf16 Al[2][64][32];
  __shared__ __align__(16) __bf16 Bl[2][64][32];
  const int tid  = threadIdx.x;
  const int lane = tid & 31;
  const int wv   = tid >> 5;
  const int rt   = wv >> 1;                // 0..3 row tile
  const int chf  = wv & 1;                 // 0..1 col half (32 cols)
  const int row0b = blockIdx.x * 64;
  const int col0b = blockIdx.y * 64;
  const int m  = lane & 15;
  const int kb = (lane < 16) ? 0 : 8;

  v8f acc0, acc1;
  #pragma unroll
  for (int i = 0; i < 8; ++i) { acc0[i] = 0.f; acc1[i] = 0.f; }

  stage_tiles(xn, wq, row0b, col0b, 0, Al[0], Bl[0], tid);
  int buf = 0;
  for (int ks = 0; ks < CC / 32; ++ks) {
    const bool more = (ks + 1 < CC / 32);
    if (more) stage_tiles(xn, wq, row0b, col0b, (ks + 1) * 32,
                          Al[buf ^ 1], Bl[buf ^ 1], tid);
    if (more) WAIT_ASYNC_2(); else WAIT_ASYNC_0();
    __syncthreads();
    const __bf16* ap  = &Al[buf][rt * 16 + m][kb];
    const __bf16* bp0 = &Bl[buf][chf * 32 + m][kb];
    const __bf16* bp1 = &Bl[buf][chf * 32 + 16 + m][kb];
    v16bf a  = cat8(*(const v8bf*)ap,  *(const v8bf*)(ap  + 16));
    v16bf b0 = cat8(*(const v8bf*)bp0, *(const v8bf*)(bp0 + 16));
    v16bf b1 = cat8(*(const v8bf*)bp1, *(const v8bf*)(bp1 + 16));
    acc0 = __builtin_amdgcn_wmma_f32_16x16x32_bf16(false, a, false, b0,
                                                   (short)0, acc0, false, false);
    acc1 = __builtin_amdgcn_wmma_f32_16x16x32_bf16(false, a, false, b1,
                                                   (short)0, acc1, false, false);
    __syncthreads();
    buf ^= 1;
  }

  // Epilogue: scatter both 16x16 tiles into head-major q/k/v.
  const int nn = lane & 15;
  #pragma unroll
  for (int t = 0; t < 2; ++t) {
    const int colt  = col0b + chf * 32 + t * 16;
    const int which = colt / CC;
    const int hh    = (colt % CC) / CE;
    const int eb    = colt % CE;
    __bf16* dst = (which == 0) ? qd : ((which == 1) ? kd : vd);
    const v8f& acc = t ? acc1 : acc0;
    #pragma unroll
    for (int i = 0; i < 8; ++i) {
      int mm = i + ((lane >> 4) << 3);
      int r  = row0b + rt * 16 + mm;
      int ni = r / CS, si = r % CS;
      dst[(((size_t)(ni * CNH + hh)) * CS + si) * CE + eb + nn] = (__bf16)acc[i];
    }
  }
}

// ---------------------------------------------------------------------------
// K4: per-head cosine norm + RoPE on q,k (wave per row; lane: {e, e+32})
// ---------------------------------------------------------------------------
__global__ __launch_bounds__(128)
void k_qknorm_rope(__bf16* __restrict__ qd, __bf16* __restrict__ kd,
                   const float* __restrict__ pos, const float* __restrict__ scale,
                   const float* __restrict__ freqs)
{
  const int lane = threadIdx.x & 31;
  const int wv   = threadIdx.x >> 5;
  const int rid  = blockIdx.x * 4 + wv;
  const int ni   = rid / (CNH * CS);
  const int h    = (rid / CS) % CNH;
  const int si   = rid % CS;
  const int b    = ni / CT;

  const size_t pbase = (((size_t)b * CH + si / CW) * CW + (si % CW)) * 2;
  const float p0 = pos[pbase + 0];
  const float p1 = pos[pbase + 1];
  const int   j  = lane & 15;
  const float fr = freqs[h * 8 + (j & 7)];
  const float th = (j < 8) ? p0 * fr : p1 * fr;
  const float ct = __cosf(th), st = __sinf(th);
  const float sq = sqrtf(scale[h]);

  __bf16* rows[2];
  rows[0] = qd + ((size_t)(ni * CNH + h) * CS + si) * CE;
  rows[1] = kd + ((size_t)(ni * CNH + h) * CS + si) * CE;

  #pragma unroll
  for (int mtx = 0; mtx < 2; ++mtx) {
    __bf16* row = rows[mtx];
    float f0 = (float)row[lane];
    float f1 = (float)row[lane + 32];
    float ss = f0 * f0 + f1 * f1;
    #pragma unroll
    for (int dd = 16; dd >= 1; dd >>= 1) ss += __shfl_xor(ss, dd, 32);
    float scl = sq * rsqrtf(ss + 1e-6f);
    f0 *= scl; f1 *= scl;
    float other = __shfl_xor(f0, 16, 32);
    float r0 = (lane < 16) ? (f0 * ct - other * st)
                           : (f0 * ct + other * st);
    row[lane]      = (__bf16)r0;
    row[lane + 32] = (__bf16)f1;
  }
}

// ---------------------------------------------------------------------------
// K5: attention; one wave per 16 query rows.  LDS-lean version (~23.5 KB):
//  - running row-max kept in registers during the S loop (shfl reductions)
//  - score strip stored as f16, overwritten in place by bf16 probabilities
//  - V chunks staged via TDM (tensor_load_to_lds + s_wait_tensorcnt)
// ---------------------------------------------------------------------------
__global__ __launch_bounds__(32)
void k_attn(const __bf16* __restrict__ qd, const __bf16* __restrict__ kd,
            const __bf16* __restrict__ vd, __bf16* __restrict__ od)
{
  __shared__ __align__(16) unsigned short SP[16][592];  // f16 scores -> bf16 P
  __shared__ __align__(16) __bf16 Vlds[32][68];         // V chunk (TDM pad)
  __shared__ float rmax[16];
  __shared__ float lsum[16];

  const int lane = threadIdx.x & 31;
  const int hb = blockIdx.x / 36;
  const int qb = blockIdx.x % 36;
  const int ni = hb / CNH, h = hb % CNH;
  const size_t base = ((size_t)(ni * CNH + h)) * CS * CE;
  const __bf16* Qb = qd + base;
  const __bf16* Kb = kd + base;
  const __bf16* Vb = vd + base;
  const int q0 = qb * 16;
  const int m    = lane & 15;
  const int half = lane >> 4;
  const int kb   = half ? 8 : 0;

  // Q fragments covering e = 0..63
  v16bf aq[2];
  #pragma unroll
  for (int c2 = 0; c2 < 2; ++c2) {
    const __bf16* ap = Qb + (size_t)(q0 + m) * CE + c2 * 32 + kb;
    aq[c2] = cat8(*(const v8bf*)ap, *(const v8bf*)(ap + 16));
  }

  // S = Q . K^T (16 x 576); track per-row running max in registers.
  float vmax[8];
  #pragma unroll
  for (int i = 0; i < 8; ++i) vmax[i] = -3.4e38f;

  for (int kt = 0; kt < CS; kt += 16) {
    if (kt + 16 < CS) __builtin_prefetch(Kb + (size_t)(kt + 16) * CE, 0, 3);
    v8f s;
    #pragma unroll
    for (int i = 0; i < 8; ++i) s[i] = 0.f;
    #pragma unroll
    for (int c2 = 0; c2 < 2; ++c2) {
      const __bf16* bp = Kb + (size_t)(kt + m) * CE + c2 * 32 + kb;
      v16bf bfr = cat8(*(const v8bf*)bp, *(const v8bf*)(bp + 16));
      s = __builtin_amdgcn_wmma_f32_16x16x32_bf16(false, aq[c2], false, bfr,
                                                  (short)0, s, false, false);
    }
    #pragma unroll
    for (int i = 0; i < 8; ++i) {
      vmax[i] = fmaxf(vmax[i], s[i]);
      SP[i + half * 8][kt + m] = f2h_bits(s[i]);   // C/D row = vgpr + 8*half
    }
  }
  // Reduce vmax across the 16 lanes of each half (rows half*8 + i).
  #pragma unroll
  for (int i = 0; i < 8; ++i) {
    float v = vmax[i];
    #pragma unroll
    for (int dd = 8; dd >= 1; dd >>= 1) v = fmaxf(v, __shfl_xor(v, dd, 32));
    vmax[i] = v;
  }
  if (m == 0) {
    #pragma unroll
    for (int i = 0; i < 8; ++i) rmax[half * 8 + i] = vmax[i];
  }
  __syncthreads();

  // P = exp(S - max) written in place (f16 -> bf16); row sums via shfl.
  {
    const int row = m;             // lanes 0..15 row r half 0; 16..31 half 1
    const float mx = rmax[row];
    float l = 0.f;
    for (int c = half * 288; c < half * 288 + 288; ++c) {
      float ex = __expf(h_bits2f(SP[row][c]) - mx);
      l += ex;
      SP[row][c] = f2bf_bits(ex);
    }
    l += __shfl_xor(l, 16, 32);
    if (half == 0) lsum[row] = l;
  }
  __syncthreads();

  // O = P . V : stage V chunk (32 x 64) in LDS, reuse P fragment for 4 e-tiles
  v8f acc[4];
  #pragma unroll
  for (int et = 0; et < 4; ++et)
    #pragma unroll
    for (int i = 0; i < 8; ++i) acc[et][i] = 0.f;

  for (int kc = 0; kc < CS; kc += 32) {
#ifdef HAVE_TDM
    asm volatile("s_wait_dscnt 0x0" ::: "memory");  // prior LDS reads done
    {
      unsigned vbase =
          (unsigned)(uintptr_t)(__attribute__((address_space(3))) __bf16*)&Vlds[0][0];
      // data_size=2B (code 1) | pad_enable | pad_interval=32dw (4) | pad=2dw (1)
      tdm_load_2d(vbase, Vb + (size_t)kc * CE,
                  /*tensor_d0=*/CE, /*tensor_d1=*/CS - kc,
                  /*tile_d0=*/CE, /*tile_d1=*/32, /*stride0=*/CE,
                  (1u << 16) | (1u << 20) | (4u << 22) | (1u << 25));
      __builtin_amdgcn_s_wait_tensorcnt(0);
      asm volatile("" ::: "memory");
    }
#else
    #pragma unroll
    for (int i = 0; i < 16; ++i) {           // 512 x 8B chunks over 32 lanes
      int cid = lane + 32 * i;
      int row = cid >> 4;
      int off = (cid & 15) * 4;
      *(v4bf*)&Vlds[row][off] =
          *(const v4bf*)(Vb + (size_t)(kc + row) * CE + off);
    }
    asm volatile("" ::: "memory");
#endif
    const __bf16* pp = (const __bf16*)&SP[m][kc + kb];
    v16bf a = cat8(*(const v8bf*)pp, *(const v8bf*)(pp + 16));
    #pragma unroll
    for (int et = 0; et < 4; ++et) {
      v16bf bfr;
      const int e = et * 16 + m;
      #pragma unroll
      for (int i = 0; i < 8; ++i) {
        bfr[i]     = Vlds[kb + i][e];
        bfr[8 + i] = Vlds[kb + 16 + i][e];
      }
      acc[et] = __builtin_amdgcn_wmma_f32_16x16x32_bf16(false, a, false, bfr,
                                                        (short)0, acc[et],
                                                        false, false);
    }
  }

  #pragma unroll
  for (int et = 0; et < 4; ++et)
    #pragma unroll
    for (int i = 0; i < 8; ++i) {
      int mm = i + half * 8;
      int r  = ni * CS + q0 + mm;
      od[(size_t)r * CC + h * CE + et * 16 + m] = (__bf16)(acc[et][i] / lsum[mm]);
    }
}

// ---------------------------------------------------------------------------
// K6: out projection + residual (same GEMM core; epilogue scatters to NCTHW)
// ---------------------------------------------------------------------------
__global__ __launch_bounds__(256)
void k_out_gemm(const __bf16* __restrict__ o, const __bf16* __restrict__ ow,
                const float* __restrict__ x, float* __restrict__ out)
{
  __shared__ __align__(16) __bf16 Al[2][64][32];
  __shared__ __align__(16) __bf16 Bl[2][64][32];
  const int tid  = threadIdx.x;
  const int lane = tid & 31;
  const int wv   = tid >> 5;
  const int rt   = wv >> 1;
  const int chf  = wv & 1;
  const int row0b = blockIdx.x * 64;
  const int col0b = blockIdx.y * 64;
  const int m  = lane & 15;
  const int kb = (lane < 16) ? 0 : 8;

  v8f acc0, acc1;
  #pragma unroll
  for (int i = 0; i < 8; ++i) { acc0[i] = 0.f; acc1[i] = 0.f; }

  stage_tiles(o, ow, row0b, col0b, 0, Al[0], Bl[0], tid);
  int buf = 0;
  for (int ks = 0; ks < CC / 32; ++ks) {
    const bool more = (ks + 1 < CC / 32);
    if (more) stage_tiles(o, ow, row0b, col0b, (ks + 1) * 32,
                          Al[buf ^ 1], Bl[buf ^ 1], tid);
    if (more) WAIT_ASYNC_2(); else WAIT_ASYNC_0();
    __syncthreads();
    const __bf16* ap  = &Al[buf][rt * 16 + m][kb];
    const __bf16* bp0 = &Bl[buf][chf * 32 + m][kb];
    const __bf16* bp1 = &Bl[buf][chf * 32 + 16 + m][kb];
    v16bf a  = cat8(*(const v8bf*)ap,  *(const v8bf*)(ap  + 16));
    v16bf b0 = cat8(*(const v8bf*)bp0, *(const v8bf*)(bp0 + 16));
    v16bf b1 = cat8(*(const v8bf*)bp1, *(const v8bf*)(bp1 + 16));
    acc0 = __builtin_amdgcn_wmma_f32_16x16x32_bf16(false, a, false, b0,
                                                   (short)0, acc0, false, false);
    acc1 = __builtin_amdgcn_wmma_f32_16x16x32_bf16(false, a, false, b1,
                                                   (short)0, acc1, false, false);
    __syncthreads();
    buf ^= 1;
  }

  const int nn = lane & 15;
  #pragma unroll
  for (int t = 0; t < 2; ++t) {
    const int colt = col0b + chf * 32 + t * 16;
    const v8f& acc = t ? acc1 : acc0;
    #pragma unroll
    for (int i = 0; i < 8; ++i) {
      int mm = i + ((lane >> 4) << 3);
      int r  = row0b + rt * 16 + mm;
      int ni = r / CS, si = r % CS;
      int b  = ni / CT, tt = ni % CT;
      int j  = colt + nn;
      size_t oi = (((size_t)b * CC + j) * CT + tt) * CS + si;
      out[oi] = acc[i] + x[oi];
    }
  }
}

// ---------------------------------------------------------------------------
// Host-side launcher.  Inputs (all f32):
//   0:x 1:pos 2:cond 3:norm_w 4:qkv_w 5:scale 6:out_w 7:freqs
// ---------------------------------------------------------------------------
extern "C" void kernel_launch(void* const* d_in, const int* in_sizes, int n_in,
                              void* d_out, int out_size, void* d_ws, size_t ws_size,
                              hipStream_t stream) {
  const float* x      = (const float*)d_in[0];
  const float* pos    = (const float*)d_in[1];
  const float* cond   = (const float*)d_in[2];
  const float* norm_w = (const float*)d_in[3];
  const float* qkv_w  = (const float*)d_in[4];
  const float* scale  = (const float*)d_in[5];
  const float* out_w  = (const float*)d_in[6];
  const float* freqs  = (const float*)d_in[7];
  float* out = (float*)d_out;

  // ws layout: cs | xn | q | k | v | o | qkv_w(bf16) | out_w(bf16)
  char* ws = (char*)d_ws;
  const size_t NE = (size_t)CROWS * CC;        // 7,077,888
  float*  cs  = (float*)ws;
  __bf16* xn  = (__bf16*)(ws + 4096);
  __bf16* qd  = xn + NE;
  __bf16* kd  = qd + NE;
  __bf16* vd  = kd + NE;
  __bf16* od  = vd + NE;
  __bf16* qwb = od + NE;
  __bf16* owb = qwb + (size_t)CQKV * CC;

  k_cvt_w<<<dim3(((unsigned)(CQKV * CC) + 255) / 256), dim3(256), 0, stream>>>(
      qkv_w, out_w, qwb, owb);
  k_cond_scale<<<dim3((CB * CC + 255) / 256), dim3(256), 0, stream>>>(
      cond, norm_w, cs);
  k_rmsnorm<<<dim3(CROWS), dim3(256), 0, stream>>>(x, cs, xn);
  k_qkv_gemm<<<dim3(CROWS / 64, CQKV / 64), dim3(256), 0, stream>>>(
      xn, qwb, qd, kd, vd);
  k_qknorm_rope<<<dim3((CN * CNH * CS) / 4), dim3(128), 0, stream>>>(
      qd, kd, pos, scale, freqs);
  k_attn<<<dim3(CN * CNH * (CS / 16)), dim3(32), 0, stream>>>(qd, kd, vd, od);
  k_out_gemm<<<dim3(CROWS / 64, CC / 64), dim3(256), 0, stream>>>(
      od, owb, x, out);
}
